// MultiSelfAttention_70832600646316
// MI455X (gfx1250) — matrix operand order repaired and checked by reading, not compiled, not run on hardware
//
#include <hip/hip_runtime.h>
#include <math.h>

// ---------------- types ----------------
typedef __attribute__((ext_vector_type(16))) __bf16 bf16x16;
typedef __attribute__((ext_vector_type(8)))  float  f32x8;
typedef __attribute__((ext_vector_type(4)))  float  f32x4;
typedef __attribute__((ext_vector_type(4)))  unsigned int u32x4;
typedef __attribute__((ext_vector_type(2)))  unsigned int u32x2;

union FragB16 { bf16x16 f; u32x4 q[2]; };

// ---------------- problem constants ----------------
constexpr int E  = 768;
constexpr int H  = 12;
constexpr int NS = 2048;
constexpr int BATCH = 4;
constexpr int MTOT = BATCH * NS;          // 8192 rows for the projections
constexpr float SCALE = 0.28867513459481287f;   // 1/sqrt(12) (per reference)

// workspace layout (bytes); all offsets multiples of 256
constexpr size_t SZ_WT  = (size_t)3 * E * E * 2;         // bf16 transposed weights
constexpr size_t OFF_WT = 0;
constexpr size_t SZ_QKV = (size_t)BATCH * NS * E * 2;    // bf16 [B][H][N][64]
constexpr size_t OFF_Q  = OFF_WT + SZ_WT;
constexpr size_t OFF_K  = OFF_Q + SZ_QKV;
constexpr size_t OFF_V  = OFF_K + SZ_QKV;
constexpr size_t SZ_MZ  = (size_t)BATCH * H * NS * 4;    // f32 per-key stats
constexpr size_t OFF_M  = OFF_V + SZ_QKV;
constexpr size_t OFF_Z  = OFF_M + SZ_MZ;

__device__ __forceinline__ unsigned short f32_to_bf16(float f) {
  unsigned int u = __float_as_uint(f);
  u += 0x7fffu + ((u >> 16) & 1u);     // round-to-nearest-even
  return (unsigned short)(u >> 16);
}
__device__ __forceinline__ unsigned int pack_bf16x2(float a, float b) {
  return (unsigned int)f32_to_bf16(a) | ((unsigned int)f32_to_bf16(b) << 16);
}

#define WMMA_BF16(A, B, C) \
  __builtin_amdgcn_wmma_f32_16x16x32_bf16(false, (A), false, (B), (short)0, (C), false, false)

// ---- gfx1250 async copy global->LDS (ASYNCcnt-tracked, no VGPR data path) ----
__device__ __forceinline__ unsigned lds_off32(const void* p) {
  // generic LDS pointer = {shared aperture, wg-relative offset} -> low 32 bits
  return (unsigned)(unsigned long long)p;
}
__device__ __forceinline__ void async_cp16(unsigned lds_dst, const void* gsrc) {
  unsigned long long ga = (unsigned long long)gsrc;
  asm volatile("global_load_async_to_lds_b128 %0, %1, off"
               :: "v"(lds_dst), "v"(ga) : "memory");
}
__device__ __forceinline__ void async_wait0() {
  asm volatile("s_wait_asynccnt 0x0" ::: "memory");
}

// ============================================================
// Kernel 0: convert weights fp32 -> bf16, transposed [e_out][e_in]
// ============================================================
__global__ __launch_bounds__(256) void prep_weights(
    const float* __restrict__ Wq, const float* __restrict__ Wk,
    const float* __restrict__ Wv, unsigned short* __restrict__ wt) {
  int i = blockIdx.x * 256 + threadIdx.x;
  if (i >= 3 * E * E) return;
  int z = i / (E * E);
  int rem = i - z * E * E;
  int r = rem / E;      // e_out
  int c = rem - r * E;  // e_in
  const float* W = (z == 0) ? Wq : (z == 1) ? Wk : Wv;
  wt[i] = f32_to_bf16(W[(size_t)c * E + r]);   // W is [e_in][e_out]
}

// ============================================================
// Kernel 1: Q/K/V = x @ W + b  (bf16 WMMA, f32 accum)
// grid: (MTOT/64, E/64, 3), block 128 (4 waves)
// output bf16, head-major [B][H][N][64]
// ============================================================
__global__ __launch_bounds__(128) void qkv_gemm(
    const float* __restrict__ x, const unsigned short* __restrict__ wt,
    const float* __restrict__ bq, const float* __restrict__ bk,
    const float* __restrict__ bv,
    unsigned short* __restrict__ outq, unsigned short* __restrict__ outk,
    unsigned short* __restrict__ outv) {
  const int z = blockIdx.z;
  const unsigned short* w = wt + (size_t)z * E * E;   // [e_out][e_in] bf16
  const float* bias = (z == 0) ? bq : (z == 1) ? bk : bv;
  unsigned short* out = (z == 0) ? outq : (z == 1) ? outk : outv;

  const int m0 = blockIdx.x * 64;
  const int n0 = blockIdx.y * 64;

  __shared__ unsigned short sA[64 * 32];   // x tile, bf16 row-major [64 m][32 k]
  __shared__ unsigned short sB[64 * 32];   // Wt tile [64 n][32 k]

  const int tid = threadIdx.x;
  const int wave = tid >> 5, lane = tid & 31;
  const int lq = lane & 15;
  const bool hi = lane >= 16;
  const int oA = hi ? 8 : 0;

  f32x8 acc[4] = {};

  for (int k0 = 0; k0 < E; k0 += 32) {
    // async-stage Wt tile (bf16 flat copy, overlaps with x conversion below)
    for (int i = tid; i < 256; i += 128) {
      int r = i >> 2, c8 = (i & 3) * 8;
      async_cp16(lds_off32(&sB[r * 32 + c8]),
                 &w[(size_t)(n0 + r) * E + k0 + c8]);
    }
    // stage x (fp32 -> bf16), 512 chunks of 4 floats
    for (int i = tid; i < 512; i += 128) {
      int r = i >> 3, c4 = (i & 7) * 4;
      f32x4 xv = *(const f32x4*)&x[(size_t)(m0 + r) * E + k0 + c4];
      u32x2 pv;
      pv[0] = pack_bf16x2(xv[0], xv[1]);
      pv[1] = pack_bf16x2(xv[2], xv[3]);
      *(u32x2*)&sA[r * 32 + c4] = pv;
    }
    async_wait0();
    __syncthreads();

    FragB16 a;
    {
      const unsigned short* rowp = &sA[(wave * 16 + lq) * 32];
      a.q[0] = *(const u32x4*)(rowp + oA);
      a.q[1] = *(const u32x4*)(rowp + oA + 16);
    }
#pragma unroll
    for (int ns = 0; ns < 4; ++ns) {
      FragB16 b;
      const unsigned short* rowp = &sB[(ns * 16 + lq) * 32 + (hi ? 8 : 0)];
      b.q[0] = *(const u32x4*)(rowp);
      b.q[1] = *(const u32x4*)(rowp + 16);
      acc[ns] = WMMA_BF16(a.f, b.f, acc[ns]);
    }
    __syncthreads();
  }

  // epilogue: bias, convert, scatter to head-major layout
#pragma unroll
  for (int ns = 0; ns < 4; ++ns) {
    int e_out = n0 + ns * 16 + lq;
    float bvv = bias[e_out];
    int h = e_out >> 6, dd = e_out & 63;
#pragma unroll
    for (int r = 0; r < 8; ++r) {
      int m = m0 + wave * 16 + r + (hi ? 8 : 0);
      int bb = m >> 11;            // m / NS
      int nn = m & (NS - 1);
      size_t idx = (((size_t)bb * H + h) * NS + nn) * 64 + dd;
      out[idx] = f32_to_bf16(acc[ns][r] + bvv);
    }
  }
}

// ============================================================
// Kernel 2: per-key softmax stats over the QUERY axis.
// grid: (NS/64 key tiles, BATCH*H), block 128 (4 waves; wave owns 16 keys)
// Double-buffered async Q-tile prefetch.
// ============================================================
__global__ __launch_bounds__(128) void col_stats(
    const unsigned short* __restrict__ Qw, const unsigned short* __restrict__ Kw,
    float* __restrict__ Mv, float* __restrict__ Zv) {
  const int bh = blockIdx.y;
  const int k0 = blockIdx.x * 64;

  __shared__ unsigned short sK[64 * 64];      // [key][d]
  __shared__ unsigned short sQ[2][16 * 64];   // double-buffered [q][d]

  const int tid = threadIdx.x;
  const int wave = tid >> 5, lane = tid & 31;
  const int lq = lane & 15;
  const bool hi = lane >= 16;
  const int oA = hi ? 8 : 0;

  const size_t rowbase = (size_t)bh * NS;
  const size_t baseK = (rowbase + k0) * 64;
  // async-stage K tile + first Q tile
  for (int i = tid; i < 512; i += 128)
    async_cp16(lds_off32(&sK[i * 8]), &Kw[baseK + (size_t)i * 8]);
  async_cp16(lds_off32(&sQ[0][tid * 8]), &Qw[rowbase * 64 + (size_t)tid * 8]);
  async_wait0();
  __syncthreads();

  // B frags (K^T) constant over the q loop
  FragB16 bf[2];
  {
    const unsigned short* kr = &sK[(wave * 16 + lq) * 64];
#pragma unroll
    for (int s = 0; s < 2; ++s) {
      int base = s * 32 + (hi ? 8 : 0);
      bf[s].q[0] = *(const u32x4*)(kr + base);
      bf[s].q[1] = *(const u32x4*)(kr + base + 16);
    }
  }

  float m_run = -3.0e38f, z_run = 0.0f;

  for (int qt = 0; qt < NS / 16; ++qt) {
    const int cur = qt & 1;
    // prefetch next Q tile into the other buffer (readers of it are
    // already past the previous barrier)
    if (qt + 1 < NS / 16)
      async_cp16(lds_off32(&sQ[cur ^ 1][tid * 8]),
                 &Qw[(rowbase + (size_t)(qt + 1) * 16) * 64 + (size_t)tid * 8]);

    FragB16 a0, a1;
    const unsigned short* qr = &sQ[cur][lq * 64];
    a0.q[0] = *(const u32x4*)(qr + oA);
    a0.q[1] = *(const u32x4*)(qr + oA + 16);
    a1.q[0] = *(const u32x4*)(qr + 32 + oA);
    a1.q[1] = *(const u32x4*)(qr + 32 + oA + 16);

    f32x8 acc = {};
    acc = WMMA_BF16(a0.f, bf[0].f, acc);
    acc = WMMA_BF16(a1.f, bf[1].f, acc);

    // online (max, sumexp) along q, per lane (lane owns one key column)
    float tm = acc[0] * SCALE;
#pragma unroll
    for (int r = 1; r < 8; ++r) tm = fmaxf(tm, acc[r] * SCALE);
    float mn = fmaxf(m_run, tm);
    float zs = 0.0f;
#pragma unroll
    for (int r = 0; r < 8; ++r) zs += __expf(acc[r] * SCALE - mn);
    z_run = z_run * __expf(m_run - mn) + zs;
    m_run = mn;

    async_wait0();     // next tile landed
    __syncthreads();   // everyone done reading current tile
  }

  // combine the two lanes (rows 0..7 / 8..15 of each tile) holding same key
  float m_o = __shfl_xor(m_run, 16, 32);
  float z_o = __shfl_xor(z_run, 16, 32);
  float mn = fmaxf(m_run, m_o);
  float zt = z_run * __expf(m_run - mn) + z_o * __expf(m_o - mn);
  if (lane < 16) {
    size_t idx = rowbase + k0 + wave * 16 + lq;
    Mv[idx] = mn;
    Zv[idx] = zt;
  }
}

// ============================================================
// Kernel 3: out = softmax_q(S) @ V, recomputing S tiles.
// grid: (NS/64 q tiles, BATCH*H), block 128 (4 waves; wave owns 16 q rows)
// ============================================================
constexpr int VT_PITCH = 40;  // padded pitch (shorts) for transposed V tile

__global__ __launch_bounds__(128) void attn_out(
    const unsigned short* __restrict__ Qw, const unsigned short* __restrict__ Kw,
    const unsigned short* __restrict__ Vw,
    const float* __restrict__ Mv, const float* __restrict__ Zv,
    float* __restrict__ out) {
  const int bh = blockIdx.y;
  const int b = bh / H, h = bh - b * H;
  const int q0 = blockIdx.x * 64;

  __shared__ unsigned short sQ[64 * 64];          // [q][d]
  __shared__ unsigned short sK[32 * 64];          // [key][d]
  __shared__ unsigned short sVt[64 * VT_PITCH];   // [d][key], padded
  __shared__ unsigned short sP[4][16 * 32];       // per-wave P tile [q][key]

  const int tid = threadIdx.x;
  const int wave = tid >> 5, lane = tid & 31;
  const int lq = lane & 15;
  const bool hi = lane >= 16;
  const int oA = hi ? 8 : 0;
  const int oB = hi ? 8 : 0;

  // async-stage this block's 64 Q rows once
  const size_t baseQ = ((size_t)bh * NS + q0) * 64;
  for (int i = tid; i < 512; i += 128)
    async_cp16(lds_off32(&sQ[i * 8]), &Qw[baseQ + (size_t)i * 8]);
  async_wait0();
  __syncthreads();

  FragB16 a0, a1;  // Q A-fragments, constant over the key loop
  {
    const unsigned short* qr = &sQ[(wave * 16 + lq) * 64];
    a0.q[0] = *(const u32x4*)(qr + oA);
    a0.q[1] = *(const u32x4*)(qr + oA + 16);
    a1.q[0] = *(const u32x4*)(qr + 32 + oA);
    a1.q[1] = *(const u32x4*)(qr + 32 + oA + 16);
  }

  f32x8 ov[4] = {};
  const float* mrow = Mv + (size_t)bh * NS;
  const float* zrow = Zv + (size_t)bh * NS;

  for (int kt = 0; kt < NS / 32; ++kt) {
    __syncthreads();
    const int kbase = kt * 32;
    const size_t gK = ((size_t)bh * NS + kbase) * 64;
    // async-stage K tile (flat copy; overlaps with V transpose below)
    for (int i = tid; i < 256; i += 128)
      async_cp16(lds_off32(&sK[i * 8]), &Kw[gK + (size_t)i * 8]);
    // stage V tile transposed: sVt[d][k]
    for (int i = tid; i < 2048; i += 128) {
      int k = i >> 6, d = i & 63;
      sVt[d * VT_PITCH + k] = Vw[gK + (size_t)k * 64 + d];
    }
    async_wait0();
    __syncthreads();

    // S = Q K^T for this wave's 16 q rows x 32 keys
    f32x8 sacc[2] = {};
#pragma unroll
    for (int j = 0; j < 2; ++j) {
      FragB16 b0, b1;
      const unsigned short* kr = &sK[(j * 16 + lq) * 64];
      b0.q[0] = *(const u32x4*)(kr + oB);
      b0.q[1] = *(const u32x4*)(kr + oB + 16);
      b1.q[0] = *(const u32x4*)(kr + 32 + oB);
      b1.q[1] = *(const u32x4*)(kr + 32 + oB + 16);
      sacc[j] = WMMA_BF16(a0.f, b0.f, sacc[j]);
      sacc[j] = WMMA_BF16(a1.f, b1.f, sacc[j]);
    }

    // P = exp(scale*S - m[k]) / Z[k], write bf16 tile to LDS
    unsigned short* pp = &sP[wave][0];
#pragma unroll
    for (int j = 0; j < 2; ++j) {
      int key = kbase + j * 16 + lq;
      float mval = mrow[key];
      float zinv = 1.0f / zrow[key];
#pragma unroll
      for (int r = 0; r < 8; ++r) {
        float p = __expf(sacc[j][r] * SCALE - mval) * zinv;
        int row = r + (hi ? 8 : 0);
        pp[row * 32 + j * 16 + lq] = f32_to_bf16(p);
      }
    }

    // re-fragment P as A operand (intra-wave LDS round trip)
    FragB16 pa;
    {
      const unsigned short* pr = &sP[wave][lq * 32];
      pa.q[0] = *(const u32x4*)(pr + oA);
      pa.q[1] = *(const u32x4*)(pr + oA + 16);
    }
    // out += P @ V
#pragma unroll
    for (int ns = 0; ns < 4; ++ns) {
      FragB16 bv;
      const unsigned short* vr = &sVt[(ns * 16 + lq) * VT_PITCH + oB];
      bv.q[0] = *(const u32x4*)(vr);
      bv.q[1] = *(const u32x4*)(vr + 16);
      ov[ns] = WMMA_BF16(pa.f, bv.f, ov[ns]);
    }
  }

  // write fp32 output, (B,N,E) layout with e = h*64 + d
#pragma unroll
  for (int ns = 0; ns < 4; ++ns) {
#pragma unroll
    for (int r = 0; r < 8; ++r) {
      int ql = q0 + wave * 16 + r + (hi ? 8 : 0);
      size_t idx = ((size_t)b * NS + ql) * E + h * 64 + ns * 16 + lq;
      out[idx] = ov[ns][r];
    }
  }
}

// ============================================================
extern "C" void kernel_launch(void* const* d_in, const int* in_sizes, int n_in,
                              void* d_out, int out_size, void* d_ws, size_t ws_size,
                              hipStream_t stream) {
  const float* x  = (const float*)d_in[0];
  const float* Wq = (const float*)d_in[1];
  const float* bq = (const float*)d_in[2];
  const float* Wk = (const float*)d_in[3];
  const float* bk = (const float*)d_in[4];
  const float* Wv = (const float*)d_in[5];
  const float* bv = (const float*)d_in[6];

  char* ws = (char*)d_ws;
  unsigned short* wt = (unsigned short*)(ws + OFF_WT);
  unsigned short* Qb = (unsigned short*)(ws + OFF_Q);
  unsigned short* Kb = (unsigned short*)(ws + OFF_K);
  unsigned short* Vb = (unsigned short*)(ws + OFF_V);
  float* Mv = (float*)(ws + OFF_M);
  float* Zv = (float*)(ws + OFF_Z);

  prep_weights<<<(3 * E * E + 255) / 256, 256, 0, stream>>>(Wq, Wk, Wv, wt);
  qkv_gemm<<<dim3(MTOT / 64, E / 64, 3), 128, 0, stream>>>(
      x, wt, bq, bk, bv, Qb, Kb, Vb);
  col_stats<<<dim3(NS / 64, BATCH * H), 128, 0, stream>>>(Qb, Kb, Mv, Zv);
  attn_out<<<dim3(NS / 64, BATCH * H), 128, 0, stream>>>(
      Qb, Kb, Vb, Mv, Zv, (float*)d_out);
}
